// TransformerBlock_78606491451653
// MI455X (gfx1250) — matrix operand order repaired and checked by reading, not compiled, not run on hardware
//
#include <hip/hip_runtime.h>
#include <hip/hip_bf16.h>
#include <math.h>

// ---------------------------------------------------------------------------
// Transformer block (prenorm) for gfx1250 / MI455X, bf16 WMMA pipeline.
//   S=4096, D=768, H=12, DK=64, F=3072
// ---------------------------------------------------------------------------

#define S_LEN 4096
#define D_MODEL 768
#define N_HEADS 12
#define DK 64
#define D_FFN 3072
#define EPS_RMS 1e-5f

typedef __attribute__((ext_vector_type(16))) __bf16 bf16x16;
typedef __attribute__((ext_vector_type(8)))  __bf16 bf16x8;
typedef __attribute__((ext_vector_type(8)))  float  f32x8;

// ---------------------------------------------------------------------------
// WMMA fragment loaders (layouts per CDNA5 ISA 7.12.2, 16-bit, wave32)
// ---------------------------------------------------------------------------

// A-matrix 16x32 bf16: lane L -> row = L%16; half = L/16.
// elements 0..7  = K in [half*8,     half*8+8)
// elements 8..15 = K in [16+half*8, 16+half*8+8)
__device__ inline bf16x16 load_a_frag(const __bf16* __restrict__ base, int ld) {
    const int lane = threadIdx.x & 31;
    const int row  = lane & 15;
    const int half = lane >> 4;
    const __bf16* p = base + (size_t)row * ld + half * 8;
    bf16x8 lo = *(const bf16x8*)(p);
    bf16x8 hi = *(const bf16x8*)(p + 16);
    return __builtin_shufflevector(lo, hi, 0,1,2,3,4,5,6,7,8,9,10,11,12,13,14,15);
}

// B-matrix 32x16 bf16: lane L -> column n = L%16; kh = L/16.
// elements 0..7  = K in [kh*8,     kh*8+8)
// elements 8..15 = K in [16+kh*8, 16+kh*8+8)
// "base" points at element (n=0, k=0) of a row-major [N,K] matrix whose row n
// is column n of B  (i.e. computes X @ W^T directly).
__device__ inline bf16x16 load_b_frag(const __bf16* __restrict__ base, int ld) {
    const int lane = threadIdx.x & 31;
    const int n  = lane & 15;
    const int kh = lane >> 4;
    const __bf16* p = base + (size_t)n * ld + kh * 8;
    bf16x8 lo = *(const bf16x8*)(p);
    bf16x8 hi = *(const bf16x8*)(p + 16);
    return __builtin_shufflevector(lo, hi, 0,1,2,3,4,5,6,7,8,9,10,11,12,13,14,15);
}

__device__ inline f32x8 wmma_bf16(bf16x16 a, bf16x16 b, f32x8 c) {
    return __builtin_amdgcn_wmma_f32_16x16x32_bf16(false, a, false, b, (short)0, c,
                                                   false, false);
}

// ---------------------------------------------------------------------------
// Elementwise helpers
// ---------------------------------------------------------------------------

__global__ void cvt_bf16_kernel(const float* __restrict__ in, __bf16* __restrict__ out, int n) {
    for (int i = blockIdx.x * blockDim.x + threadIdx.x; i < n; i += gridDim.x * blockDim.x)
        out[i] = (__bf16)in[i];
}

// v [S, D] bf16  ->  vt [D, S] bf16   (per-head-dim-major so PV B-frags are contiguous)
__global__ void vtr_kernel(const __bf16* __restrict__ v, __bf16* __restrict__ vt) {
    int i = blockIdx.x * blockDim.x + threadIdx.x;
    if (i < S_LEN * D_MODEL) {
        int s = i / D_MODEL;
        int c = i % D_MODEL;
        vt[(size_t)c * S_LEN + s] = v[i];
    }
}

// rmsnorm row kernel: out_bf16[row] = x[row]*g * rsqrt(mean(x^2)+eps)
__global__ __launch_bounds__(256)
void rmsnorm_kernel(const float* __restrict__ x, const float* __restrict__ g,
                    __bf16* __restrict__ out) {
    const int row = blockIdx.x;
    const float* xr = x + (size_t)row * D_MODEL;
    float v[3];
    float ss = 0.f;
#pragma unroll
    for (int i = 0; i < 3; ++i) {
        v[i] = xr[threadIdx.x + i * 256];
        ss += v[i] * v[i];
    }
#pragma unroll
    for (int off = 16; off > 0; off >>= 1) ss += __shfl_xor(ss, off, 32);
    __shared__ float wsum[8];
    const int wave = threadIdx.x >> 5;
    if ((threadIdx.x & 31) == 0) wsum[wave] = ss;
    __syncthreads();
    float tot = 0.f;
#pragma unroll
    for (int i = 0; i < 8; ++i) tot += wsum[i];
    const float inv = rsqrtf(tot * (1.0f / D_MODEL) + EPS_RMS);
#pragma unroll
    for (int i = 0; i < 3; ++i) {
        int c = threadIdx.x + i * 256;
        out[(size_t)row * D_MODEL + c] = (__bf16)(v[i] * g[c] * inv);
    }
}

// ---------------------------------------------------------------------------
// GEMM:  C[M,N] = A[M,K](bf16) @ W[N,K](bf16)^T  with epilogue variants.
// Block: 256 threads = 8 waves (2x4), block tile 64x256, wave tile 32x64.
// ---------------------------------------------------------------------------

enum { EPI_BF16 = 0, EPI_RESID = 1, EPI_GELU = 2 };

template <int EPI>
__global__ __launch_bounds__(256)
void gemm_wmma(const __bf16* __restrict__ A, const __bf16* __restrict__ W,
               __bf16* __restrict__ outB, float* __restrict__ outF,
               const float* __restrict__ resid,
               int M, int N, int K, float scale) {
    const int wave = threadIdx.x >> 5;
    const int wm = wave & 1;
    const int wn = wave >> 1;
    const int m0 = blockIdx.x * 64 + wm * 32;
    const int n0 = blockIdx.y * 256 + wn * 64;

    f32x8 c[2][4] = {};

    for (int k0 = 0; k0 < K; k0 += 32) {
        bf16x16 a0 = load_a_frag(A + (size_t)m0 * K + k0, K);
        bf16x16 a1 = load_a_frag(A + (size_t)(m0 + 16) * K + k0, K);
#pragma unroll
        for (int j = 0; j < 4; ++j) {
            bf16x16 b = load_b_frag(W + (size_t)(n0 + j * 16) * K + k0, K);
            c[0][j] = wmma_bf16(a0, b, c[0][j]);
            c[1][j] = wmma_bf16(a1, b, c[1][j]);
        }
        if (k0 + 32 < K) {  // hint next K-slab into cache (global_prefetch_b8)
            __builtin_prefetch(A + (size_t)m0 * K + k0 + 32, 0, 1);
            __builtin_prefetch(W + (size_t)n0 * K + k0 + 32, 0, 1);
        }
    }

    // Epilogue. C/D layout: lanes 0-15 -> rows 0..7 (vgpr r), col = lane;
    //                       lanes 16-31 -> rows 8..15, col = lane-16.
    const int lane = threadIdx.x & 31;
    const int col  = lane & 15;
    const int half = lane >> 4;
#pragma unroll
    for (int i = 0; i < 2; ++i) {
#pragma unroll
        for (int j = 0; j < 4; ++j) {
#pragma unroll
            for (int r = 0; r < 8; ++r) {
                const int m = m0 + i * 16 + half * 8 + r;
                const int n = n0 + j * 16 + col;
                const float v = c[i][j][r];
                const size_t idx = (size_t)m * N + n;
                if constexpr (EPI == EPI_BF16) {
                    outB[idx] = (__bf16)(v * scale);
                } else if constexpr (EPI == EPI_RESID) {
                    outF[idx] = v + resid[idx];
                } else {  // exact GELU: 0.5*x*(1+erf(x/sqrt(2)))
                    const float gl = 0.5f * v * (1.0f + erff(v * 0.70710678118654752f));
                    outB[idx] = (__bf16)gl;
                }
            }
        }
    }
}

// ---------------------------------------------------------------------------
// Flash attention, causal. One wave per (16-row query tile, head).
// q is pre-scaled by 1/sqrt(DK). vt is [D_MODEL, S] (head-dim-major V^T).
// ---------------------------------------------------------------------------

__global__ __launch_bounds__(256)
void attn_kernel(const __bf16* __restrict__ q, const __bf16* __restrict__ k,
                 const __bf16* __restrict__ vt, __bf16* __restrict__ o) {
    __shared__ alignas(64) __bf16 pT[8][16][32];  // per-wave P tile staging

    const int wave = threadIdx.x >> 5;
    const int lane = threadIdx.x & 31;
    const int col  = lane & 15;
    const int half = lane >> 4;

    const int qt = blockIdx.x * 8 + wave;   // query tile (0..255)
    const int h  = blockIdx.y;              // head
    const int q0 = qt * 16;

    const __bf16* qbase = q + (size_t)q0 * D_MODEL + h * DK;
    const bf16x16 aq0 = load_a_frag(qbase + 0,  D_MODEL);
    const bf16x16 aq1 = load_a_frag(qbase + 32, D_MODEL);

    f32x8 acc[4] = {};
    float mrow[8], lrow[8];
#pragma unroll
    for (int r = 0; r < 8; ++r) { mrow[r] = -1e30f; lrow[r] = 0.f; }

    for (int kk = 0; kk < q0 + 16; kk += 32) {
        // --- S = Q K^T for 32 keys (two 16-key tiles) ---
        f32x8 s0 = {}, s1 = {};
        {
            const __bf16* kb0 = k + (size_t)kk * D_MODEL + h * DK;
            s0 = wmma_bf16(aq0, load_b_frag(kb0 + 0,  D_MODEL), s0);
            s0 = wmma_bf16(aq1, load_b_frag(kb0 + 32, D_MODEL), s0);
            const __bf16* kb1 = k + (size_t)(kk + 16) * D_MODEL + h * DK;
            s1 = wmma_bf16(aq0, load_b_frag(kb1 + 0,  D_MODEL), s1);
            s1 = wmma_bf16(aq1, load_b_frag(kb1 + 32, D_MODEL), s1);
        }

        // --- causal mask + online softmax stats (keys live on lanes) ---
#pragma unroll
        for (int r = 0; r < 8; ++r) {
            const int qg  = q0 + half * 8 + r;
            const int kg0 = kk + col;
            const int kg1 = kk + 16 + col;
            const float v0 = (kg0 <= qg) ? s0[r] : -1e30f;
            const float v1 = (kg1 <= qg) ? s1[r] : -1e30f;

            float mx = fmaxf(v0, v1);
#pragma unroll
            for (int off = 1; off < 16; off <<= 1) mx = fmaxf(mx, __shfl_xor(mx, off, 32));
            const float mnew = fmaxf(mrow[r], mx);
            const float sc = __expf(mrow[r] - mnew);
            const float p0 = __expf(v0 - mnew);
            const float p1 = __expf(v1 - mnew);
            float rs = p0 + p1;
#pragma unroll
            for (int off = 1; off < 16; off <<= 1) rs += __shfl_xor(rs, off, 32);
            lrow[r] = lrow[r] * sc + rs;
            mrow[r] = mnew;
#pragma unroll
            for (int j = 0; j < 4; ++j) acc[j][r] *= sc;

            pT[wave][half * 8 + r][col]      = (__bf16)p0;
            pT[wave][half * 8 + r][16 + col] = (__bf16)p1;
        }
        asm volatile("s_wait_dscnt 0" ::: "memory");

        // --- reload P in A-fragment layout (16x32) ---
        bf16x8 lo = *(const bf16x8*)(&pT[wave][col][half * 8]);
        bf16x8 hi = *(const bf16x8*)(&pT[wave][col][16 + half * 8]);
        bf16x16 pa = __builtin_shufflevector(lo, hi, 0,1,2,3,4,5,6,7,8,9,10,11,12,13,14,15);

        // --- O += P @ V : B columns are head dims, K = 32 keys (contiguous in vt) ---
#pragma unroll
        for (int j = 0; j < 4; ++j) {
            const __bf16* vb = vt + ((size_t)(h * DK + j * 16)) * S_LEN + kk;
            acc[j] = wmma_bf16(pa, load_b_frag(vb, S_LEN), acc[j]);
        }
    }

    // --- normalize and store ---
#pragma unroll
    for (int j = 0; j < 4; ++j) {
        const float linv = 1.0f;  // per-row divide below
#pragma unroll
        for (int r = 0; r < 8; ++r) {
            const float ov = acc[j][r] / lrow[r];
            const int row = q0 + half * 8 + r;
            o[(size_t)row * D_MODEL + h * DK + j * 16 + col] = (__bf16)ov;
        }
        (void)linv;
    }
}

// ---------------------------------------------------------------------------
// Host launcher
// ---------------------------------------------------------------------------

extern "C" void kernel_launch(void* const* d_in, const int* in_sizes, int n_in,
                              void* d_out, int out_size, void* d_ws, size_t ws_size,
                              hipStream_t stream) {
    (void)in_sizes; (void)n_in; (void)out_size; (void)ws_size;

    const float* x  = (const float*)d_in[0];
    const float* g1 = (const float*)d_in[1];
    const float* g2 = (const float*)d_in[2];
    const float* wq = (const float*)d_in[3];
    const float* wk = (const float*)d_in[4];
    const float* wv = (const float*)d_in[5];
    const float* wo = (const float*)d_in[6];
    const float* w1 = (const float*)d_in[7];
    const float* w2 = (const float*)d_in[8];
    float* out = (float*)d_out;

    // workspace carve
    char* base = (char*)d_ws;
    size_t off = 0;
    auto carve = [&](size_t bytes) {
        void* p = base + off;
        off = (off + bytes + 255) & ~(size_t)255;
        return p;
    };
    const size_t DD = (size_t)D_MODEL * D_MODEL;
    const size_t SD = (size_t)S_LEN * D_MODEL;
    const size_t FD = (size_t)D_FFN * D_MODEL;
    const size_t SF = (size_t)S_LEN * D_FFN;

    __bf16* wqb = (__bf16*)carve(DD * 2);
    __bf16* wkb = (__bf16*)carve(DD * 2);
    __bf16* wvb = (__bf16*)carve(DD * 2);
    __bf16* wob = (__bf16*)carve(DD * 2);
    __bf16* w1b = (__bf16*)carve(FD * 2);
    __bf16* w2b = (__bf16*)carve(FD * 2);
    __bf16* xnb = (__bf16*)carve(SD * 2);
    __bf16* qb  = (__bf16*)carve(SD * 2);
    __bf16* kb  = (__bf16*)carve(SD * 2);
    __bf16* vb  = (__bf16*)carve(SD * 2);
    __bf16* vtb = (__bf16*)carve(SD * 2);
    __bf16* ob  = (__bf16*)carve(SD * 2);
    float*  yf  = (float*) carve(SD * 4);
    __bf16* hb  = (__bf16*)carve(SD * 2);
    __bf16* f1b = (__bf16*)carve(SF * 2);

    const int TB = 256;
    auto cvt = [&](const float* src, __bf16* dst, size_t n) {
        cvt_bf16_kernel<<<(int)((n + TB - 1) / TB), TB, 0, stream>>>(src, dst, (int)n);
    };

    // 1) weights -> bf16
    cvt(wq, wqb, DD); cvt(wk, wkb, DD); cvt(wv, wvb, DD); cvt(wo, wob, DD);
    cvt(w1, w1b, FD); cvt(w2, w2b, FD);

    // 2) rmsnorm(x, g1) -> bf16
    rmsnorm_kernel<<<S_LEN, TB, 0, stream>>>(x, g1, xnb);

    // 3) QKV projections (q pre-scaled by 1/sqrt(DK) = 0.125)
    dim3 gproj(S_LEN / 64, D_MODEL / 256);
    gemm_wmma<EPI_BF16><<<gproj, TB, 0, stream>>>(xnb, wqb, qb, nullptr, nullptr,
                                                  S_LEN, D_MODEL, D_MODEL, 0.125f);
    gemm_wmma<EPI_BF16><<<gproj, TB, 0, stream>>>(xnb, wkb, kb, nullptr, nullptr,
                                                  S_LEN, D_MODEL, D_MODEL, 1.0f);
    gemm_wmma<EPI_BF16><<<gproj, TB, 0, stream>>>(xnb, wvb, vb, nullptr, nullptr,
                                                  S_LEN, D_MODEL, D_MODEL, 1.0f);

    // 4) V^T per head-dim for contiguous PV B-fragments
    vtr_kernel<<<(int)((SD + TB - 1) / TB), TB, 0, stream>>>(vb, vtb);

    // 5) causal flash attention -> o (bf16)
    attn_kernel<<<dim3(S_LEN / 16 / 8, N_HEADS), TB, 0, stream>>>(qb, kb, vtb, ob);

    // 6) out projection + residual -> y (f32)
    gemm_wmma<EPI_RESID><<<gproj, TB, 0, stream>>>(ob, wob, nullptr, yf, x,
                                                   S_LEN, D_MODEL, D_MODEL, 1.0f);

    // 7) rmsnorm(y, g2) -> h (bf16)
    rmsnorm_kernel<<<S_LEN, TB, 0, stream>>>(yf, g2, hb);

    // 8) FFN up + exact GELU -> f1 (bf16)
    dim3 gup(S_LEN / 64, D_FFN / 256);
    gemm_wmma<EPI_GELU><<<gup, TB, 0, stream>>>(hb, w1b, f1b, nullptr, nullptr,
                                                S_LEN, D_FFN, D_MODEL, 1.0f);

    // 9) FFN down + residual(y) -> out (f32)
    gemm_wmma<EPI_RESID><<<gproj, TB, 0, stream>>>(f1b, w2b, nullptr, out, yf,
                                                   S_LEN, D_MODEL, D_FFN, 1.0f);
}
